// SparseConv_40819369181593
// MI455X (gfx1250) — matrix operand order repaired and checked by reading, not compiled, not run on hardware
//
#include <hip/hip_runtime.h>
#include <stdint.h>

// Sparse 3x3x3 voxel conv as a WMMA GEMM:
//   out[m,f] = sum_k w[m,k] * feat[nbr[m,k],:] @ Kcell(k)[:,f] + bias[f]
// cell(k) = fixed digit-reversal of k (voxel-center geometry => fcoord=rel+1);
// masked neighbors redirected to a zero feature row.
//
// Pipeline:
//   prep_weights : kernel f32 -> bf16, pre-swizzled into WMMA B-fragment layout (ws)
//   prep_feats   : feats * importance -> bf16 rows (+1 zero row) (ws)
//   sparse_conv  : async-copy weights ws->LDS, then 27 x (K=32) WMMA steps,
//                  2 M-tiles x 4 N-tiles per wave (8 wmma / k-iter).

typedef __attribute__((ext_vector_type(16))) __bf16 v16bf;
typedef __attribute__((ext_vector_type(8)))  float  v8f;

#define CIN   32
#define FOUT  64
#define KNB   27
#define ROWS_PER_BLK   256          // 8 waves x 32 rows
#define WDWORDS (KNB * 4 * 32 * 8)  // 27648 dwords
#define WBYTES  (WDWORDS * 4)       // 110592 B = LDS weight cache

static __device__ __forceinline__ uint32_t pack_bf16x2(float a, float b) {
    union { __bf16 h[2]; uint32_t u; } p;
    p.h[0] = (__bf16)a;
    p.h[1] = (__bf16)b;
    return p.u;
}

// ws weight layout == B-fragment register image:
// byte offset = cell*4096 + t*1024 + lane*32 + j*4
// lane n (0-15): col f=t*16+n, K=c=0..15 (2/dword); lane n+16: c=16..31.
__global__ __launch_bounds__(256) void prep_weights(
    const float* __restrict__ kern, uint32_t* __restrict__ wsW)
{
    int ld   = blockIdx.x * 256 + threadIdx.x;   // grid covers WDWORDS exactly
    int j    = ld & 7;
    int ln   = (ld >> 3) & 31;
    int t    = (ld >> 8) & 3;
    int cell = ld >> 10;
    int n    = ln & 15;
    int chal = ln >> 4;
    int c    = chal * 16 + 2 * j;
    int f    = t * 16 + n;
    const float* kp = kern + cell * (CIN * FOUT) + c * FOUT + f;
    wsW[ld] = pack_bf16x2(kp[0], kp[FOUT]);
}

// feat'[i,c] = bf16(feats[i,c] * importance[i]); row nIn is all-zero (mask sink).
__global__ __launch_bounds__(256) void prep_feats(
    const float* __restrict__ feats, const float* __restrict__ imp,
    uint32_t* __restrict__ wsF, int nIn)
{
    int i = blockIdx.x * 256 + threadIdx.x;      // one dword = 2 channels
    int total = (nIn + 1) * (CIN / 2);
    if (i >= total) return;
    int pt = i >> 4;
    uint32_t v = 0;
    if (pt < nIn) {
        float w = imp[pt];
        v = pack_bf16x2(feats[2 * i] * w, feats[2 * i + 1] * w);
    }
    wsF[i] = v;
}

__global__ __launch_bounds__(256) void sparse_conv_wmma(
    const uint32_t*      __restrict__ wsW,      // bf16 weights, B-frag layout
    const char*          __restrict__ wsF,      // bf16 feats [nIn+1][64 B]
    const float*         __restrict__ bias,     // [64]
    const int*           __restrict__ nbr_idx,  // [M,27]
    const unsigned char* __restrict__ nbr_mask, // [M,27]
    float*               __restrict__ out,      // [M,64]
    int M, int nIn)
{
    extern __shared__ uint32_t ldsk[];
    const int tid = threadIdx.x;

    // ---- Async global->LDS copy of the weight cache (ASYNCcnt-tracked) ----
    {
        uint32_t lbase = (uint32_t)(uintptr_t)(&ldsk[0]) + tid * 16;
        uint64_t gbase = (uint64_t)(uintptr_t)wsW + tid * 16;
        #pragma unroll
        for (int i = 0; i < WBYTES / (256 * 16); ++i) {   // 27 iters
            uint32_t l = lbase + i * 4096;
            uint64_t g = gbase + i * 4096;
            asm volatile("global_load_async_to_lds_b128 %0, %1, off"
                         :: "v"(l), "v"(g) : "memory");
        }
        asm volatile("s_wait_asynccnt 0x0" ::: "memory");
    }
    __syncthreads();

    const int lane = tid & 31;
    const int wave = tid >> 5;
    const int m0   = blockIdx.x * ROWS_PER_BLK + wave * 32;
    const int n    = lane & 15;
    const int half = lane >> 4;

    // Two A row-tiles per wave; clamp rows to keep EXEC all-ones for WMMA.
    int r0 = m0 + n;        r0 = r0 < M ? r0 : (M - 1);
    int r1 = m0 + 16 + n;   r1 = r1 < M ? r1 : (M - 1);

    v8f acc[2][4] = {};

    for (int k = 0; k < KNB; ++k) {
        const int cell = (k % 3) * 9 + ((k / 3) % 3) * 3 + (k / 9);

        int q0 = nbr_idx[r0 * KNB + k];
        int q1 = nbr_idx[r1 * KNB + k];
        q0 = nbr_mask[r0 * KNB + k] ? q0 : nIn;   // masked -> zero row
        q1 = nbr_mask[r1 * KNB + k] ? q1 : nIn;

        // A 16x32 bf16 layout: lanes 0-15 carry c in {0..15},
        // lanes 16-31 carry c in {16..31}; pairs already packed in ws.
        const uint4* p0 = (const uint4*)(wsF + (size_t)q0 * 64);
        const uint4* p1 = (const uint4*)(wsF + (size_t)q1 * 64);
        union { uint4 q[2]; v16bf v; } A0, A1;
        A0.q[0] = p0[2 * half];     A0.q[1] = p0[2 * half + 1];
        A1.q[0] = p1[2 * half];     A1.q[1] = p1[2 * half + 1];

        const char* bptr = (const char*)ldsk + cell * 4096 + lane * 32;
        #pragma unroll
        for (int t = 0; t < 4; ++t) {
            v16bf b = *(const v16bf*)(bptr + t * 1024);
            acc[0][t] = __builtin_amdgcn_wmma_f32_16x16x32_bf16(
                false, A0.v, false, b, (short)0, acc[0][t], false, false);
            acc[1][t] = __builtin_amdgcn_wmma_f32_16x16x32_bf16(
                false, A1.v, false, b, (short)0, acc[1][t], false, false);
        }
    }

    // ---- Epilogue: C/D layout -> VGPR j holds row j (lanes 0-15) / 8+j ----
    float bb[4];
    #pragma unroll
    for (int t = 0; t < 4; ++t) bb[t] = bias[t * 16 + n];

    #pragma unroll
    for (int s = 0; s < 2; ++s) {
        #pragma unroll
        for (int j = 0; j < 8; ++j) {
            int m = m0 + s * 16 + half * 8 + j;
            if (m < M) {
                float* op = out + (size_t)m * FOUT + n;
                #pragma unroll
                for (int t = 0; t < 4; ++t) op[t * 16] = acc[s][t][j] + bb[t];
            }
        }
    }
}

extern "C" void kernel_launch(void* const* d_in, const int* in_sizes, int n_in,
                              void* d_out, int out_size, void* d_ws, size_t ws_size,
                              hipStream_t stream) {
    // setup_inputs() order:
    // 0 inp_features f32   1 inp_positions (unused)  2 out_positions (unused)
    // 3 voxel_size (unused) 4 inp_importance f32     5 kernel f32
    // 6 bias f32            7 neighbor_index i32     8 neighbor_mask bool(u8)
    const float*         feats = (const float*)d_in[0];
    const float*         imp   = (const float*)d_in[4];
    const float*         kern  = (const float*)d_in[5];
    const float*         bias  = (const float*)d_in[6];
    const int*           nbr   = (const int*)d_in[7];
    const unsigned char* msk   = (const unsigned char*)d_in[8];
    float*               out   = (float*)d_out;

    const int M   = in_sizes[7] / KNB;
    const int nIn = in_sizes[4];

    // Scratch: [0, WBYTES) bf16 weights (B-frag layout); then bf16 feature rows
    // (nIn+1 rows x 64 B, last row zero). Requires ws_size >= ~9.71 MB.
    uint32_t* wsW = (uint32_t*)d_ws;
    uint32_t* wsF = (uint32_t*)((char*)d_ws + WBYTES);

    prep_weights<<<WDWORDS / 256, 256, 0, stream>>>(kern, wsW);
    prep_feats<<<((nIn + 1) * (CIN / 2) + 255) / 256, 256, 0, stream>>>(
        feats, imp, wsF, nIn);
    sparse_conv_wmma<<<(M + ROWS_PER_BLK - 1) / ROWS_PER_BLK, 256, WBYTES, stream>>>(
        wsW, (const char*)wsF, bias, nbr, msk, out, M, nIn);
}
